// Net_11914239279183
// MI455X (gfx1250) — matrix operand order repaired and checked by reading, not compiled, not run on hardware
//
#include <hip/hip_runtime.h>

typedef __attribute__((ext_vector_type(16))) _Float16 v16h;
typedef __attribute__((ext_vector_type(8)))  float    v8f;

static __device__ __forceinline__ float sigmoidf_(float x) {
    return 1.0f / (1.0f + __expf(-x));
}

// ---------------------------------------------------------------------------
// WMMA GEMM: C[M x NC] = A[M x K] * B (+ bias), fp32 in/out, fp16 MACs.
//   TRANS_B == false : B is [K x NC] row-major (B[k*NC + n])
//   TRANS_B == true  : B is [NC x K] row-major (A * B^T), contiguous in k
// One wave computes one 16x16 tile via v_wmma_f32_16x16x32_f16, K-step 32.
// REQUIRES: M % 16 == 0, NC % 16 == 0, K % 32 == 0  ->  no bounds checks,
// all A / trans-B fragment loads are unguarded 16B vector loads.
// ---------------------------------------------------------------------------
template <bool TRANS_B, bool ADD_BIAS>
__global__ __launch_bounds__(256) void wmma_gemm_f16(
    const float* __restrict__ A, const float* __restrict__ B,
    const float* __restrict__ bias, float* __restrict__ C,
    int M, int K, int NC)
{
    const int tilesN = NC >> 4;
    const int tilesM = M >> 4;
    const int wid = (blockIdx.x * blockDim.x + threadIdx.x) >> 5;
    if (wid >= tilesM * tilesN) return;          // whole-wave uniform exit
    const int mt = wid / tilesN;
    const int nt = wid - mt * tilesN;
    const int lane   = threadIdx.x & 31;
    const int rowA   = lane & 15;
    const int colB   = lane & 15;
    const int khalfA = (lane >> 4) << 4;         // A: K-chunk base 0 or 16
    const int kofsB  = (lane >> 4) << 4;         // B: K-chunk base 0 or 16
    const int colg   = nt * 16 + colB;

    const float* Arow = A + (size_t)(mt * 16 + rowA) * K;
    v8f acc = {};

    for (int kk = 0; kk < K; kk += 32) {
        // ---- A fragment: 16 contiguous floats per lane ----
        const float* ap = Arow + kk + khalfA;    // 64B-aligned (K%32==0)
        float4 x0 = *reinterpret_cast<const float4*>(ap);
        float4 x1 = *reinterpret_cast<const float4*>(ap + 4);
        float4 x2 = *reinterpret_cast<const float4*>(ap + 8);
        float4 x3 = *reinterpret_cast<const float4*>(ap + 12);
        v16h a;
        a[0]  = (_Float16)x0.x; a[1]  = (_Float16)x0.y;
        a[2]  = (_Float16)x0.z; a[3]  = (_Float16)x0.w;
        a[4]  = (_Float16)x1.x; a[5]  = (_Float16)x1.y;
        a[6]  = (_Float16)x1.z; a[7]  = (_Float16)x1.w;
        a[8]  = (_Float16)x2.x; a[9]  = (_Float16)x2.y;
        a[10] = (_Float16)x2.z; a[11] = (_Float16)x2.w;
        a[12] = (_Float16)x3.x; a[13] = (_Float16)x3.y;
        a[14] = (_Float16)x3.z; a[15] = (_Float16)x3.w;

        // ---- B fragment: lanes 0-15 hold K 0..15, lanes 16-31 K 16..31 ----
        v16h b;
        if constexpr (TRANS_B) {
            const float* bp = B + (size_t)colg * K + kk + kofsB; // 64B-aligned
            float4 y0 = *reinterpret_cast<const float4*>(bp);
            float4 y1 = *reinterpret_cast<const float4*>(bp + 4);
            float4 y2 = *reinterpret_cast<const float4*>(bp + 8);
            float4 y3 = *reinterpret_cast<const float4*>(bp + 12);
            b[0]  = (_Float16)y0.x; b[1]  = (_Float16)y0.y;
            b[2]  = (_Float16)y0.z; b[3]  = (_Float16)y0.w;
            b[4]  = (_Float16)y1.x; b[5]  = (_Float16)y1.y;
            b[6]  = (_Float16)y1.z; b[7]  = (_Float16)y1.w;
            b[8]  = (_Float16)y2.x; b[9]  = (_Float16)y2.y;
            b[10] = (_Float16)y2.z; b[11] = (_Float16)y2.w;
            b[12] = (_Float16)y3.x; b[13] = (_Float16)y3.y;
            b[14] = (_Float16)y3.z; b[15] = (_Float16)y3.w;
        } else {
            const float* bp = B + (size_t)(kk + kofsB) * NC + colg;
#pragma unroll
            for (int e = 0; e < 16; ++e)
                b[e] = (_Float16)bp[(size_t)e * NC];  // coalesced across lanes
        }

        acc = __builtin_amdgcn_wmma_f32_16x16x32_f16(
                  false, a, false, b, (short)0, acc, false, false);
    }

    const int mhi = (lane >> 4) << 3;            // C: lanes16-31 hold M=8..15
    float bv = ADD_BIAS ? bias[colg] : 0.0f;
#pragma unroll
    for (int i = 0; i < 8; ++i) {
        int rowg = mt * 16 + mhi + i;
        C[(size_t)rowg * NC + colg] = acc[i] + bv;
    }
}

// ---------------------------------------------------------------------------
// GCN helpers
// ---------------------------------------------------------------------------
__global__ void k_fill(float* __restrict__ p, float v, int n) {
    int i = blockIdx.x * blockDim.x + threadIdx.x;
    if (i < n) p[i] = v;
}

__global__ void k_deg(const int* __restrict__ dst, float* __restrict__ deg, int E) {
    int e = blockIdx.x * blockDim.x + threadIdx.x;
    if (e < E) atomicAdd(&deg[dst[e]], 1.0f);
}

__global__ void k_rsqrt(float* __restrict__ p, int n) {
    int i = blockIdx.x * blockDim.x + threadIdx.x;
    if (i < n) p[i] = rsqrtf(p[i]);             // deg >= 1 always (self-loop)
}

template <int F>
__global__ void k_selfinit(const float* __restrict__ h, const float* __restrict__ dinv,
                           float* __restrict__ out, int total) {
    int idx = blockIdx.x * blockDim.x + threadIdx.x;
    if (idx >= total) return;
    int i = idx / F;                            // F is power of two -> shift
    float d = dinv[i];
    out[idx] = h[idx] * d * d;                  // self-loop contribution
}

template <int F>
__global__ void k_edge(const int* __restrict__ src, const int* __restrict__ dst,
                       const float* __restrict__ dinv, const float* __restrict__ h,
                       float* __restrict__ out, int total) {
    int idx = blockIdx.x * blockDim.x + threadIdx.x;
    if (idx >= total) return;
    int e = idx / F;
    int f = idx - e * F;
    int s = src[e], d = dst[e];
    atomicAdd(&out[(size_t)d * F + f], h[(size_t)s * F + f] * dinv[s] * dinv[d]);
}

template <int F>
__global__ void k_biasrelu(float* __restrict__ p, const float* __restrict__ b, int total) {
    int idx = blockIdx.x * blockDim.x + threadIdx.x;
    if (idx >= total) return;
    int f = idx % F;
    float v = p[idx] + b[f];
    p[idx] = v > 0.0f ? v : 0.0f;
}

// pad [N,16] -> [N,32] (upper 16 features zero), so conv2 GEMM gets K=32
__global__ void k_pad_h(const float* __restrict__ h16, float* __restrict__ h32, int total) {
    int idx = blockIdx.x * blockDim.x + threadIdx.x;   // total = N*32
    if (idx >= total) return;
    int i = idx >> 5, c = idx & 31;
    h32[idx] = (c < 16) ? h16[(size_t)i * 16 + c] : 0.0f;
}

// pad conv2_w [16,32] -> [32,32] with zero rows 16..31
__global__ void k_pad_w(const float* __restrict__ w, float* __restrict__ wp) {
    int idx = blockIdx.x * blockDim.x + threadIdx.x;   // 1024 threads
    if (idx >= 32 * 32) return;
    int r = idx >> 5;
    wp[idx] = (r < 16) ? w[idx] : 0.0f;
}

// ---------------------------------------------------------------------------
// Persistent GRU scan. block 0 = forward, block 1 = reverse (separate WGPs).
// W_hh (192x64) lives in LDS with stride-65 padding (conflict-free: bank =
// (tid + k) & 63). Hidden state h[64] in LDS; 4 accumulators for FMA ILP.
// torch gate order: rows 0..63 = r, 64..127 = z, 128..191 = n.
// ---------------------------------------------------------------------------
__global__ __launch_bounds__(192) void gru_scan(
    const float* __restrict__ gx_f, const float* __restrict__ gx_r,
    const float* __restrict__ whh_f, const float* __restrict__ bhh_f,
    const float* __restrict__ whh_r, const float* __restrict__ bhh_r,
    float* __restrict__ out_f, float* __restrict__ out_r, int N)
{
    const bool fwd = (blockIdx.x == 0);
    const float* gx  = fwd ? gx_f  : gx_r;
    const float* whh = fwd ? whh_f : whh_r;
    const float* bhh = fwd ? bhh_f : bhh_r;
    float* out = fwd ? out_f : out_r;
    const int tid = threadIdx.x;

    __shared__ float w[192 * 65];
    __shared__ float h[64];
    __shared__ float g[192];

    for (int i = tid; i < 192 * 64; i += 192) {
        int r = i >> 6, c = i & 63;
        w[r * 65 + c] = whh[i];
    }
    if (tid < 64) h[tid] = 0.0f;
    const float bh = bhh[tid];
    __syncthreads();

    const float* wr = &w[tid * 65];
    for (int s = 0; s < N; ++s) {
        const int t = fwd ? s : (N - 1 - s);
        float a0 = 0.f, a1 = 0.f, a2 = 0.f, a3 = 0.f;
#pragma unroll
        for (int k = 0; k < 64; k += 4) {
            a0 += wr[k]     * h[k];
            a1 += wr[k + 1] * h[k + 1];
            a2 += wr[k + 2] * h[k + 2];
            a3 += wr[k + 3] * h[k + 3];
        }
        g[tid] = bh + ((a0 + a1) + (a2 + a3));
        __syncthreads();                         // g ready; h reads done
        if (tid < 64) {
            const float* gxt = gx + (size_t)t * 192;
            float r  = sigmoidf_(gxt[tid]        + g[tid]);
            float z  = sigmoidf_(gxt[64 + tid]   + g[64 + tid]);
            float nc = tanhf(    gxt[128 + tid]  + r * g[128 + tid]);
            float hn = (1.0f - z) * nc + z * h[tid];
            h[tid] = hn;
            out[(size_t)t * 64 + tid] = hn;
        }
        __syncthreads();                         // h ready for next matvec
    }
}

__global__ void k_concat(const float* __restrict__ f, const float* __restrict__ r,
                         float* __restrict__ o, int total /* N*64 */) {
    int idx = blockIdx.x * blockDim.x + threadIdx.x;
    if (idx >= total) return;
    int i = idx >> 6, j = idx & 63;
    o[(size_t)i * 128 + j]      = f[idx];
    o[(size_t)i * 128 + 64 + j] = r[idx];
}

__global__ void k_final(const float* __restrict__ f1, const float* __restrict__ r1,
                        const float* __restrict__ lw, const float* __restrict__ lb,
                        float* __restrict__ out, int total /* N*10 */) {
    int idx = blockIdx.x * blockDim.x + threadIdx.x;
    if (idx >= total) return;
    int i = idx / 10;
    int c = idx - i * 10;
    const float* wf = lw + (size_t)c * 128;
    float acc = lb[c];
#pragma unroll 8
    for (int j = 0; j < 64; ++j) acc += f1[(size_t)i * 64 + j] * wf[j];
#pragma unroll 8
    for (int j = 0; j < 64; ++j) acc += r1[(size_t)i * 64 + j] * wf[64 + j];
    out[idx] = acc;
}

// ---------------------------------------------------------------------------
extern "C" void kernel_launch(void* const* d_in, const int* in_sizes, int n_in,
                              void* d_out, int out_size, void* d_ws, size_t ws_size,
                              hipStream_t stream) {
    (void)n_in; (void)out_size; (void)ws_size;
    const float* x        = (const float*)d_in[0];
    const int*   ei       = (const int*)d_in[1];
    const float* conv1_w  = (const float*)d_in[2];
    const float* conv1_b  = (const float*)d_in[3];
    const float* conv2_w  = (const float*)d_in[4];
    const float* conv2_b  = (const float*)d_in[5];
    const float* w_ih_l0  = (const float*)d_in[6];
    const float* w_hh_l0  = (const float*)d_in[7];
    const float* b_ih_l0  = (const float*)d_in[8];
    const float* b_hh_l0  = (const float*)d_in[9];
    const float* w_ih_l0r = (const float*)d_in[10];
    const float* w_hh_l0r = (const float*)d_in[11];
    const float* b_ih_l0r = (const float*)d_in[12];
    const float* b_hh_l0r = (const float*)d_in[13];
    const float* w_ih_l1  = (const float*)d_in[14];
    const float* w_hh_l1  = (const float*)d_in[15];
    const float* b_ih_l1  = (const float*)d_in[16];
    const float* b_hh_l1  = (const float*)d_in[17];
    const float* w_ih_l1r = (const float*)d_in[18];
    const float* w_hh_l1r = (const float*)d_in[19];
    const float* b_ih_l1r = (const float*)d_in[20];
    const float* b_hh_l1r = (const float*)d_in[21];
    const float* lin_w    = (const float*)d_in[22];
    const float* lin_b    = (const float*)d_in[23];

    const int N = in_sizes[0] / 128;             // 50000 (divisible by 16)
    const int E = in_sizes[1] / 2;               // 1.6M
    const int* src = ei;
    const int* dst = ei + E;

    // workspace carve (floats): N*(1+32+32+192+192+64+64+128)+1024 ~= 141 MB
    float* ws = (float*)d_ws;
    size_t o = 0;
    float* dinv = ws + o; o += (size_t)N;
    float* bufA = ws + o; o += (size_t)N * 32;
    float* bufB = ws + o; o += (size_t)N * 32;
    float* gxf  = ws + o; o += (size_t)N * 192;
    float* gxr  = ws + o; o += (size_t)N * 192;
    float* seqf = ws + o; o += (size_t)N * 64;
    float* seqr = ws + o; o += (size_t)N * 64;
    float* h1   = ws + o; o += (size_t)N * 128;
    float* wpad = ws + o; o += 1024;

    auto cdiv = [](long a, long b) { return (int)((a + b - 1) / b); };
    const int TB = 256;

    // ---- degree / norm ----
    k_fill<<<cdiv(N, TB), TB, 0, stream>>>(dinv, 1.0f, N);
    k_deg<<<cdiv(E, TB), TB, 0, stream>>>(dst, dinv, E);
    k_rsqrt<<<cdiv(N, TB), TB, 0, stream>>>(dinv, N);

    // ---- GCN conv1: [N,128]x[128,16] -> bufA[N,16] ----
    wmma_gemm_f16<false, false><<<cdiv((long)(N / 16) * 1, 8), TB, 0, stream>>>(
        x, conv1_w, nullptr, bufA, N, 128, 16);
    k_selfinit<16><<<cdiv((long)N * 16, TB), TB, 0, stream>>>(bufA, dinv, bufB, N * 16);
    k_edge<16><<<cdiv((long)E * 16, TB), TB, 0, stream>>>(src, dst, dinv, bufA, bufB, E * 16);
    k_biasrelu<16><<<cdiv((long)N * 16, TB), TB, 0, stream>>>(bufB, conv1_b, N * 16);

    // ---- zero-pad to K=32 so conv2 GEMM runs guard-free ----
    k_pad_h<<<cdiv((long)N * 32, TB), TB, 0, stream>>>(bufB, bufA, N * 32);
    k_pad_w<<<cdiv(1024, TB), TB, 0, stream>>>(conv2_w, wpad);

    // ---- GCN conv2: [N,32]x[32,32] -> bufB[N,32] ----
    wmma_gemm_f16<false, false><<<cdiv((long)(N / 16) * 2, 8), TB, 0, stream>>>(
        bufA, wpad, nullptr, bufB, N, 32, 32);
    k_selfinit<32><<<cdiv((long)N * 32, TB), TB, 0, stream>>>(bufB, dinv, bufA, N * 32);
    k_edge<32><<<cdiv((long)E * 32, TB), TB, 0, stream>>>(src, dst, dinv, bufB, bufA, E * 32);
    k_biasrelu<32><<<cdiv((long)N * 32, TB), TB, 0, stream>>>(bufA, conv2_b, N * 32);

    // ---- GRU layer 0 input projections: [N,32] x W_ih^T -> [N,192] ----
    wmma_gemm_f16<true, true><<<cdiv((long)(N / 16) * 12, 8), TB, 0, stream>>>(
        bufA, w_ih_l0, b_ih_l0, gxf, N, 32, 192);
    wmma_gemm_f16<true, true><<<cdiv((long)(N / 16) * 12, 8), TB, 0, stream>>>(
        bufA, w_ih_l0r, b_ih_l0r, gxr, N, 32, 192);
    gru_scan<<<2, 192, 0, stream>>>(gxf, gxr, w_hh_l0, b_hh_l0, w_hh_l0r, b_hh_l0r,
                                    seqf, seqr, N);
    k_concat<<<cdiv((long)N * 64, TB), TB, 0, stream>>>(seqf, seqr, h1, N * 64);

    // ---- GRU layer 1: [N,128] x W_ih^T -> [N,192] ----
    wmma_gemm_f16<true, true><<<cdiv((long)(N / 16) * 12, 8), TB, 0, stream>>>(
        h1, w_ih_l1, b_ih_l1, gxf, N, 128, 192);
    wmma_gemm_f16<true, true><<<cdiv((long)(N / 16) * 12, 8), TB, 0, stream>>>(
        h1, w_ih_l1r, b_ih_l1r, gxr, N, 128, 192);
    gru_scan<<<2, 192, 0, stream>>>(gxf, gxr, w_hh_l1, b_hh_l1, w_hh_l1r, b_hh_l1r,
                                    seqf, seqr, N);

    // ---- linear head: [N,128] x [10,128]^T + b ----
    k_final<<<cdiv((long)N * 10, TB), TB, 0, stream>>>(seqf, seqr, lin_w, lin_b,
                                                       (float*)d_out, N * 10);
}
